// CausalSelfAttention_3496103379107
// MI455X (gfx1250) — matrix operand order repaired and checked by reading, not compiled, not run on hardware
//
#include <hip/hip_runtime.h>
#include <hip/hip_bf16.h>

// ---------- types ----------
typedef __bf16 bf16;
typedef __attribute__((ext_vector_type(16))) __bf16 v16bf;
typedef __attribute__((ext_vector_type(8)))  __bf16 v8bf;
typedef __attribute__((ext_vector_type(4)))  __bf16 v4bf;
typedef __attribute__((ext_vector_type(8)))  float  v8f;
typedef __attribute__((ext_vector_type(4)))  float  f32x4;
typedef unsigned int u32;
typedef __attribute__((ext_vector_type(4)))  u32    u32x4;
typedef __attribute__((ext_vector_type(8)))  u32    u32x8;

#define NEGV (-1.0e9f)
#define EPSV (1e-9f)

// Load a v16bf A/B fragment from two 16-byte aligned runs of 8 bf16.
__device__ __forceinline__ v16bf ld_frag16(const bf16* p0, const bf16* p1) {
  v8bf lo = *(const v8bf*)p0;
  v8bf hi = *(const v8bf*)p1;
  return __builtin_shufflevector(lo, hi, 0,1,2,3,4,5,6,7,8,9,10,11,12,13,14,15);
}

__device__ __forceinline__ v8f wmma_bf16(v16bf a, v16bf b, v8f c) {
  return __builtin_amdgcn_wmma_f32_16x16x32_bf16(false, a, false, b, (short)0, c,
                                                 false, false);
}

// ---------- fp32 -> bf16 conversion ----------
__global__ void cvt_f32_bf16(const float* __restrict__ in, bf16* __restrict__ out, int n4) {
  int i = blockIdx.x * blockDim.x + threadIdx.x;
  if (i < n4) {
    f32x4 f = *(const f32x4*)(in + (size_t)i * 4);
    v4bf o;
    o[0] = (bf16)f[0]; o[1] = (bf16)f[1]; o[2] = (bf16)f[2]; o[3] = (bf16)f[3];
    *(v4bf*)(out + (size_t)i * 4) = o;
  }
}

// ---------- bf16 GEMM: C(MxN) = A(MxK) * B(KxN), both row-major ----------
// Block tile 128x128, 8 waves each owning 32x64 (2x4 WMMA accs), K step 32.
template <bool OUT_BF16>
__global__ __launch_bounds__(256) void gemm_bf16_kernel(
    const bf16* __restrict__ A, const bf16* __restrict__ B,
    bf16* __restrict__ Cbf, float* __restrict__ Cf, int M, int N, int K) {
  constexpr int MT = 128, NT = 128, KT = 32;
  __shared__ bf16 As[MT * KT];   // [m][k], stride 32
  __shared__ bf16 Bts[NT * KT];  // [n][k]  (transposed), stride 32

  const int tid  = threadIdx.x;
  const int lane = tid & 31;
  const int wave = tid >> 5;
  const int wm = wave >> 1;      // 0..3 -> m sub-tile of 32
  const int wn = wave & 1;       // 0..1 -> n sub-tile of 64
  const int g  = lane >> 4;
  const int ln = lane & 15;
  const int mblk = blockIdx.x * MT;
  const int nblk = blockIdx.y * NT;

  const int a_row = tid >> 1;          // 0..127
  const int a_c0  = (tid & 1) * 16;    // 0 / 16 halves
  const int b_k   = tid >> 3;          // 0..31
  const int b_n0  = (tid & 7) * 16;    // 0..112

  v8f acc[2][4] = {};

  for (int k0 = 0; k0 < K; k0 += KT) {
    __syncthreads();
    {
      const bf16* ag = A + (size_t)(mblk + a_row) * K + (k0 + a_c0);
      *(u32x4*)&As[a_row * KT + a_c0]     = *(const u32x4*)ag;
      *(u32x4*)&As[a_row * KT + a_c0 + 8] = *(const u32x4*)(ag + 8);

      const bf16* bg = B + (size_t)(k0 + b_k) * N + (nblk + b_n0);
      v8bf b0 = __builtin_bit_cast(v8bf, *(const u32x4*)bg);
      v8bf b1 = __builtin_bit_cast(v8bf, *(const u32x4*)(bg + 8));
#pragma unroll
      for (int j = 0; j < 8; ++j) {
        Bts[(b_n0 + j) * KT + b_k]     = b0[j];
        Bts[(b_n0 + 8 + j) * KT + b_k] = b1[j];
      }
      if (k0 + KT < K) {
        __builtin_prefetch(ag + KT, 0, 3);
        __builtin_prefetch(bg + (size_t)KT * N, 0, 3);
      }
    }
    __syncthreads();

    v16bf af[2], bf4[4];
#pragma unroll
    for (int mi = 0; mi < 2; ++mi) {
      const int r = wm * 32 + mi * 16 + ln;
      af[mi] = ld_frag16(&As[r * KT + g * 8], &As[r * KT + 16 + g * 8]);
    }
#pragma unroll
    for (int ni = 0; ni < 4; ++ni) {
      const int r = wn * 64 + ni * 16 + ln;
      bf4[ni] = ld_frag16(&Bts[r * KT + g * 8], &Bts[r * KT + 16 + g * 8]);
    }
#pragma unroll
    for (int mi = 0; mi < 2; ++mi)
#pragma unroll
      for (int ni = 0; ni < 4; ++ni)
        acc[mi][ni] = wmma_bf16(af[mi], bf4[ni], acc[mi][ni]);
  }

#pragma unroll
  for (int mi = 0; mi < 2; ++mi)
#pragma unroll
    for (int ni = 0; ni < 4; ++ni) {
      const int n = nblk + wn * 64 + ni * 16 + ln;
#pragma unroll
      for (int r = 0; r < 8; ++r) {
        const int m = mblk + wm * 32 + mi * 16 + g * 8 + r;
        if (OUT_BF16) Cbf[(size_t)m * N + n] = (bf16)acc[mi][ni][r];
        else          Cf [(size_t)m * N + n] = acc[mi][ni][r];
      }
    }
}

// ---------- sliding-window GQA attention, ALiBi, online softmax ----------
// grid: (nq=64, H=16, B=2); 128 threads = 4 waves; wave w owns query rows w*16..+15.
// K tile staged via the Tensor Data Mover; V via manual transposed staging.
// Row-sum l is fused into the P*V WMMA via a constant ones-column B fragment.
__global__ __launch_bounds__(128) void attn_kernel(const bf16* __restrict__ qkv,
                                                   bf16* __restrict__ att) {
  constexpr int T = 4096, D = 64, C = 1024, ROW = 1536, Gh = 4;
  const int nqi = blockIdx.x;
  const int h   = blockIdx.y;
  const int b   = blockIdx.z;
  const int hkv = h / Gh;
  const int qbase = nqi * 64;

  __shared__ bf16 Ks[32 * 64];     // [key][d] row-major (TDM destination)
  __shared__ bf16 Vts[64 * 32];    // [d][key] transposed
  __shared__ bf16 Ps[4 * 16 * 32]; // per-wave P tile re-layout

  const int tid = threadIdx.x, wave = tid >> 5, lane = tid & 31;
  const int g = lane >> 4, ln = lane & 15;
  const float slope = exp2f(-0.5f * (float)(h + 1));
  const float scale = 0.125f;

  const int qrow = qbase + wave * 16 + ln;
  const bf16* qp = qkv + (size_t)(b * T + qrow) * ROW + h * D;
  const v16bf qa0 = ld_frag16(qp + g * 8,      qp + 16 + g * 8);
  const v16bf qa1 = ld_frag16(qp + 32 + g * 8, qp + 48 + g * 8);

  const bf16* Kg = qkv + (size_t)b * T * ROW + C + hkv * D;  // K[t][d], stride ROW
  const bf16* Vg = Kg + 4 * D;                               // V[t][d]

  // constant B-fragment with a single ones column at n==0:
  // P * ones accumulates the row-sum l in accumulator column 0.
  v16bf ones;
  {
    const bf16 val = (ln == 0) ? (bf16)1.0f : (bf16)0.0f;
#pragma unroll
    for (int e = 0; e < 16; ++e) ones[e] = val;
  }

  v8f o[4] = {};
  v8f lacc = {};               // l rides here (column 0), same recurrence as O
  float mrow[8], dist0[8];
#pragma unroll
  for (int r = 0; r < 8; ++r) {
    mrow[r] = -1.0e30f;
    // dist for the first tile (kbase = qbase-512): qg - (kbase+ln)
    dist0[r] = (float)(512 + wave * 16 + g * 8 + r - ln);
  }

  // V loader indices
  const int l_key = tid >> 2;          // 0..31
  const int l_c0  = (tid & 3) * 16;    // 0,16,32,48 halves

  const u32 lds_off = (u32)(size_t)(void*)Ks;  // low 32 bits = LDS offset

  for (int kbase = qbase - 512; kbase <= qbase + 32; kbase += 32) {
    if (kbase + 31 < 0) {
#pragma unroll
      for (int r = 0; r < 8; ++r) dist0[r] -= 32.0f;
      continue;  // uniform across block
    }
    __syncthreads();

    // ---- K tile via Tensor Data Mover: 2D tile of 8-byte elements ----
    // tile = 16 elem (128B row) x 32 rows, row stride 3072B (=384 elem)
    if (wave == 0) {
      const unsigned long long ga =
          (unsigned long long)(size_t)(Kg + (long long)kbase * ROW);
      u32x4 g0;
      g0[0] = 1u;                                   // count=1, user descriptor
      g0[1] = lds_off;                              // lds_addr
      g0[2] = (u32)ga;                              // global_addr[31:0]
      g0[3] = (u32)((ga >> 32) & 0x01FFFFFFu) | (2u << 30);  // addr[56:32], type=2
      u32x8 g1;
      g1[0] = (3u << 16);         // workgroup_mask=0, data_size=3 (8B)
      g1[1] = (16u << 16);        // tensor_dim0[15:0] = 16
      g1[2] = (32u << 16);        // tensor_dim0[31:16]=0, tensor_dim1[15:0]=32
      g1[3] = (16u << 16);        // tensor_dim1[31:16]=0, tile_dim0=16
      g1[4] = 32u;                // tile_dim1=32, tile_dim2=0
      g1[5] = 384u;               // tensor_dim0_stride[31:0] = 384 elem
      g1[6] = 0u;
      g1[7] = 0u;
      asm volatile("tensor_load_to_lds %0, %1\n\t"
                   "s_wait_tensorcnt 0x0"
                   :: "s"(g0), "s"(g1)
                   : "memory");
    }

    // ---- V tile: cooperative transposed staging ----
    {
      const int kpos = kbase + l_key;
      if (kpos >= 0) {
        const bf16* vg = Vg + (size_t)kpos * ROW + l_c0;
        v8bf v0 = __builtin_bit_cast(v8bf, *(const u32x4*)vg);
        v8bf v1 = __builtin_bit_cast(v8bf, *(const u32x4*)(vg + 8));
#pragma unroll
        for (int j = 0; j < 8; ++j) {
          Vts[(l_c0 + j) * 32 + l_key]     = v0[j];
          Vts[(l_c0 + 8 + j) * 32 + l_key] = v1[j];
        }
      } else {
#pragma unroll
        for (int j = 0; j < 16; ++j)
          Vts[(l_c0 + j) * 32 + l_key] = (bf16)0.0f;
      }
    }
    __syncthreads();

    // ---- scores: S(16q x 32key) = Q(16x64) * K^T ----
    v8f s0 = {}, s1 = {};
    {
      const v16bf k00 = ld_frag16(&Ks[ln * 64 + g * 8],       &Ks[ln * 64 + 16 + g * 8]);
      const v16bf k01 = ld_frag16(&Ks[ln * 64 + 32 + g * 8],  &Ks[ln * 64 + 48 + g * 8]);
      const v16bf k10 = ld_frag16(&Ks[(16 + ln) * 64 + g * 8],      &Ks[(16 + ln) * 64 + 16 + g * 8]);
      const v16bf k11 = ld_frag16(&Ks[(16 + ln) * 64 + 32 + g * 8], &Ks[(16 + ln) * 64 + 48 + g * 8]);
      s0 = wmma_bf16(qa0, k00, s0);
      s0 = wmma_bf16(qa1, k01, s0);
      s1 = wmma_bf16(qa0, k10, s1);
      s1 = wmma_bf16(qa1, k11, s1);
    }

    // ---- online softmax (incremental float ALiBi distance) ----
    const bool kp0ok = (kbase + ln) >= 0;
    const bool kp1ok = (kbase + 16 + ln) >= 0;
    bf16* Pw = &Ps[wave * 512];
#pragma unroll
    for (int r = 0; r < 8; ++r) {
      const float d0 = dist0[r];
      const float d1 = d0 - 16.0f;
      const bool va = (d0 >= 0.0f) & (d0 < 512.0f) & kp0ok;
      const bool vb = (d1 >= 0.0f) & (d1 < 512.0f) & kp1ok;
      float e0 = va ? fmaf(-slope, d0, s0[r] * scale) : NEGV;
      float e1 = vb ? fmaf(-slope, d1, s1[r] * scale) : NEGV;
      float mx = fmaxf(e0, e1);
#pragma unroll
      for (int msk = 1; msk < 16; msk <<= 1)
        mx = fmaxf(mx, __shfl_xor(mx, msk, 32));   // stays within 16-lane group
      const float mnew = fmaxf(mrow[r], mx);
      const float corr = __expf(mrow[r] - mnew);
      const float p0 = va ? __expf(e0 - mnew) : 0.0f;
      const float p1 = vb ? __expf(e1 - mnew) : 0.0f;
      mrow[r] = mnew;
#pragma unroll
      for (int nt = 0; nt < 4; ++nt) o[nt][r] *= corr;
      lacc[r] *= corr;
      Pw[(g * 8 + r) * 32 + ln]      = (bf16)p0;
      Pw[(g * 8 + r) * 32 + 16 + ln] = (bf16)p1;
      dist0[r] = d0 - 32.0f;
    }

    // ---- O += P(16x32) * V(32x64); l += P * ones (fused row-sum) ----
    const v16bf pa = ld_frag16(&Pw[ln * 32 + g * 8], &Pw[ln * 32 + 16 + g * 8]);
#pragma unroll
    for (int nt = 0; nt < 4; ++nt) {
      const int dn = nt * 16 + ln;
      const v16bf vbF = ld_frag16(&Vts[dn * 32 + g * 8], &Vts[dn * 32 + 16 + g * 8]);
      o[nt] = wmma_bf16(pa, vbF, o[nt]);
    }
    lacc = wmma_bf16(pa, ones, lacc);
  }

  // normalize + write bf16 attention output.
  // l for row m=g*8+r lives in lane (g*16) of lacc[r]; broadcast within group.
#pragma unroll
  for (int r = 0; r < 8; ++r) {
    const float lsum = __shfl(lacc[r], lane & 16, 32);
    const float inv = 1.0f / (lsum + EPSV);
    const int trow = qbase + wave * 16 + g * 8 + r;
    bf16* op = att + (size_t)(b * T + trow) * C + h * D;
#pragma unroll
    for (int nt = 0; nt < 4; ++nt)
      op[nt * 16 + ln] = (bf16)(o[nt][r] * inv);
  }
}

// ---------- launch ----------
extern "C" void kernel_launch(void* const* d_in, const int* in_sizes, int n_in,
                              void* d_out, int out_size, void* d_ws, size_t ws_size,
                              hipStream_t stream) {
  const float* x    = (const float*)d_in[0];  // (2,4096,1024)
  const float* wqkv = (const float*)d_in[1];  // (1024,1536)
  const float* wo   = (const float*)d_in[2];  // (1024,1024)
  float* out = (float*)d_out;                 // (2,4096,1024) f32

  const int Bb = 2, Tt = 4096, Cc = 1024, ROW = 1536;
  const size_t nx  = (size_t)Bb * Tt * Cc;
  const size_t nwq = (size_t)Cc * ROW;
  const size_t nwo = (size_t)Cc * Cc;
  const size_t nqkv = (size_t)Bb * Tt * ROW;

  char* ws = (char*)d_ws;
  bf16* xh   = (bf16*)ws;
  bf16* wqh  = (bf16*)(ws + nx * 2);
  bf16* woh  = (bf16*)(ws + nx * 2 + nwq * 2);
  bf16* qkvh = (bf16*)(ws + nx * 2 + nwq * 2 + nwo * 2);
  bf16* atth = (bf16*)(ws + nx * 2 + nwq * 2 + nwo * 2 + nqkv * 2);

  cvt_f32_bf16<<<(int)((nx / 4 + 255) / 256), 256, 0, stream>>>(x, xh, (int)(nx / 4));
  cvt_f32_bf16<<<(int)((nwq / 4 + 255) / 256), 256, 0, stream>>>(wqkv, wqh, (int)(nwq / 4));
  cvt_f32_bf16<<<(int)((nwo / 4 + 255) / 256), 256, 0, stream>>>(wo, woh, (int)(nwo / 4));

  // QKV projection: (8192 x 1024) * (1024 x 1536)
  dim3 g1(8192 / 128, ROW / 128);
  gemm_bf16_kernel<true><<<g1, 256, 0, stream>>>(xh, wqh, qkvh, nullptr, 8192, ROW, Cc);

  // sliding-window attention
  attn_kernel<<<dim3(64, 16, Bb), 128, 0, stream>>>(qkvh, atth);

  // output projection: (8192 x 1024) * (1024 x 1024)
  dim3 g2(8192 / 128, Cc / 128);
  gemm_bf16_kernel<false><<<g2, 256, 0, stream>>>(atth, woh, nullptr, out, 8192, Cc, Cc);
}
